// SpatialAttention_90082644067052
// MI455X (gfx1250) — matrix operand order repaired and checked by reading, not compiled, not run on hardware
//
#include <hip/hip_runtime.h>
#include <hip/hip_bf16.h>

typedef __attribute__((ext_vector_type(16))) _Float16 v16h;
typedef __attribute__((ext_vector_type(8)))  _Float16 v8h;
typedef __attribute__((ext_vector_type(8)))  float    v8f;
typedef __attribute__((ext_vector_type(4)))  int      v4i;

// ---------------------------------------------------------------------------
// WMMA fragment helpers (wave32; A/B k order is this code's contiguous one, not the ISA table)
// A (16x32 f16): lane 0-15 -> rows, half=lane>>4 selects K subsets.
//   element e: K = half*16 + e            (contiguous 16 halfs = 32B)
// B (32x16 f16) loaded from a [N][K]-major tile: lane col = lane&15,
//   element e: K = half*16 + e -> one contiguous 32B v16h per lane.
// C/D (16x16 f32): VGPR r -> row r + 8*half, col = lane&15.
// ---------------------------------------------------------------------------
__device__ __forceinline__ v16h load_a_frag(const _Float16* base, int ld) {
    const int lane = threadIdx.x & 31;
    const int hf   = lane >> 4;
    const int row  = lane & 15;
    const _Float16* p = base + row * ld + hf * 16;
    v8h lo = *(const v8h*)(p);
    v8h hi = *(const v8h*)(p + 8);
    v16h a;
#pragma unroll
    for (int i = 0; i < 8; ++i) { a[i] = lo[i]; a[i + 8] = hi[i]; }
    return a;
}

__device__ __forceinline__ v16h load_bt_frag(const _Float16* baseT, int ld) {
    const int lane = threadIdx.x & 31;
    const int hf   = lane >> 4;
    const int col  = lane & 15;
    return *(const v16h*)(baseT + col * ld + hf * 16);
}

__device__ __forceinline__ v8f wmma_f16(v16h a, v16h b, v8f c) {
    return __builtin_amdgcn_wmma_f32_16x16x32_f16(
        /*neg_a=*/false, a, /*neg_b=*/false, b,
        /*c_mod=*/(short)0, c, /*reuse_a=*/false, /*reuse_b=*/false);
}

// ---------------------------------------------------------------------------
// CDNA5 async global->LDS (ASYNCcnt) helpers.
// Builtin signature (from hipcc diagnostic): arg0 = addrspace(1) v4i* (global
// source), arg1 = addrspace(3) v4i* (LDS dest), then imm offset, imm cpol.
// ---------------------------------------------------------------------------
__device__ __forceinline__ void async_load_b128(void* lds_dst, const void* gsrc) {
#if __has_builtin(__builtin_amdgcn_global_load_async_to_lds_b128)
    __builtin_amdgcn_global_load_async_to_lds_b128(
        (__attribute__((address_space(1))) v4i*)(v4i*)gsrc,
        (__attribute__((address_space(3))) v4i*)(v4i*)lds_dst,
        0, 0);
#else
    unsigned a = (unsigned)(unsigned long long)
        (__attribute__((address_space(3))) void*)lds_dst;
    asm volatile("global_load_async_to_lds_b128 %0, %1, off"
                 :: "v"(a), "v"(gsrc) : "memory");
#endif
}

__device__ __forceinline__ void wait_asynccnt0() {
#if __has_builtin(__builtin_amdgcn_s_wait_asynccnt)
    __builtin_amdgcn_s_wait_asynccnt(0);
#else
    asm volatile("s_wait_asynccnt 0x0" ::: "memory");
#endif
}

// ---------------------------------------------------------------------------
// Kernel 0: f32 -> f16 convert (x and the four weight matrices)
// ---------------------------------------------------------------------------
__global__ void cvt_f16(const float* __restrict__ src, _Float16* __restrict__ dst, int n) {
    int i = blockIdx.x * blockDim.x + threadIdx.x;
    if (i < n) dst[i] = (_Float16)src[i];
}

// ---------------------------------------------------------------------------
// Kernel 1: fused QKV projection.  out[g] = x @ Wg^T + bg, stored f16 in
// [B,H,N,Dh] layout.  grid = (256/64 cols, 8192/64 rows, 3), block = 128.
// Each wave computes a 16x64 tile: 8 k-steps x 4 WMMAs.
// ---------------------------------------------------------------------------
__global__ __launch_bounds__(128) void qkv_proj(
    const _Float16* __restrict__ xh,
    const _Float16* __restrict__ WqH, const _Float16* __restrict__ WkH,
    const _Float16* __restrict__ WvH,
    const float* __restrict__ bq, const float* __restrict__ bk,
    const float* __restrict__ bv,
    _Float16* __restrict__ Qh, _Float16* __restrict__ Kh,
    _Float16* __restrict__ Vh) {
    const int wv   = threadIdx.x >> 5;
    const int lane = threadIdx.x & 31;
    const int hf   = lane >> 4;
    const int l    = lane & 15;
    const int z    = blockIdx.z;
    const _Float16* WH = (z == 0) ? WqH : (z == 1) ? WkH : WvH;
    const float* bias  = (z == 0) ? bq  : (z == 1) ? bk  : bv;
    _Float16* dst      = (z == 0) ? Qh  : (z == 1) ? Kh  : Vh;

    const int n0 = blockIdx.x * 64;            // output column tile (one head)
    const int m0 = blockIdx.y * 64 + wv * 16;  // output row tile

    v8f zero8 = {0.f, 0.f, 0.f, 0.f, 0.f, 0.f, 0.f, 0.f};
    v8f acc[4];
#pragma unroll
    for (int t = 0; t < 4; ++t) acc[t] = zero8;

#pragma unroll
    for (int s = 0; s < 8; ++s) {
        v16h a = load_a_frag(xh + m0 * 256 + s * 32, 256);
#pragma unroll
        for (int t = 0; t < 4; ++t) {
            v16h b = load_bt_frag(WH + (n0 + t * 16) * 256 + s * 32, 256);
            acc[t] = wmma_f16(a, b, acc[t]);
        }
    }

    const int h = n0 >> 6;  // head index (64-wide col tile == one head)
#pragma unroll
    for (int t = 0; t < 4; ++t) {
#pragma unroll
        for (int r = 0; r < 8; ++r) {
            int gr = m0 + r + hf * 8;   // global row 0..8191
            int c  = n0 + t * 16 + l;   // global col 0..255
            int bb = gr >> 11;          // batch
            int ns = gr & 2047;         // sequence pos
            int d  = c & 63;            // within-head dim
            float v = acc[t][r] + bias[c];
            dst[(((bb * 4 + h) * 2048) + ns) * 64 + d] = (_Float16)v;
        }
    }
}

// ---------------------------------------------------------------------------
// Kernel 2: chunked attention with per-chunk softmax (faithful to reference:
// softmax over each 256-key chunk, chunk contexts SUMMED).
// grid = (2048/64 query tiles, B*H), block = 128 (4 waves x 16 query rows).
// Q fragments are chunk-invariant -> kept in VGPRs.  The next chunk's V is
// DMA'd global->LDS with async_to_lds_b128 (ASYNCcnt) while the current
// chunk's QK^T + softmax + PV run; at the chunk boundary we wait ASYNCcnt,
// barrier, and transpose Vraw->Vt in LDS for the WMMA B-operand layout.
// ---------------------------------------------------------------------------
__global__ __launch_bounds__(128) void attn(
    const _Float16* __restrict__ Qh, const _Float16* __restrict__ Kh,
    const _Float16* __restrict__ Vh, const int* __restrict__ adj,
    _Float16* __restrict__ CtxH) {
    __shared__ _Float16 Vraw[256 * 64];  // async landing buffer: [key][d]
    __shared__ _Float16 Vt[64 * 256];    // V chunk transposed:   [d][key]
    __shared__ _Float16 Ps[64 * 256];    // unnormalized probs:   [qrow][key]

    const int tid  = threadIdx.x;
    const int wv   = tid >> 5;
    const int lane = tid & 31;
    const int hf   = lane >> 4;
    const int l    = lane & 15;
    const int bh   = blockIdx.y;       // 0..15
    const int bb   = bh >> 2;
    const int h    = bh & 3;
    const int m0   = blockIdx.x * 64;  // query row base within N

    const size_t base = (size_t)bh * 2048 * 64;
    const _Float16* Qb = Qh + base;
    const _Float16* Kb = Kh + base;
    const _Float16* Vb = Vh + base;

    // issue 32 KB async copy of chunk j's V (16B per lane per issue)
    auto issue_v = [&](int j) {
        const _Float16* src = Vb + (size_t)j * 16384;
#pragma unroll
        for (int it = 0; it < 16; ++it) {
            int idx = it * 128 + tid;                 // 16-byte packet index
            async_load_b128(&Vraw[idx * 8], src + idx * 8);
        }
    };
    auto transpose_v = [&]() {
        for (int i = tid; i < 64 * 256; i += 128) {
            int key = i & 255, d = i >> 8;
            Vt[d * 256 + key] = Vraw[key * 64 + d];
        }
    };

    issue_v(0);

    const int qrow = m0 + wv * 16;
    // Q fragments for this wave's 16 rows, Dh=64 -> two K-steps, loop-invariant
    v16h a0 = load_a_frag(Qb + qrow * 64 + 0, 64);
    v16h a1 = load_a_frag(Qb + qrow * 64 + 32, 64);

    v8f zero8 = {0.f, 0.f, 0.f, 0.f, 0.f, 0.f, 0.f, 0.f};
    v8f ctx[4];
#pragma unroll
    for (int t = 0; t < 4; ++t) ctx[t] = zero8;

    wait_asynccnt0();
    __syncthreads();
    transpose_v();
    __syncthreads();

    for (int j = 0; j < 8; ++j) {  // key chunks
        if (j < 7) {
            issue_v(j + 1);  // overlap next V DMA with this chunk's compute
            // warm L2/L0 for next chunk's adjacency columns
            __builtin_prefetch(&adj[(m0 + (tid >> 1)) * 2048 + (j + 1) * 256 +
                                    (tid & 1) * 128], 0, 1);
        }

        // ---- scores: 16 column tiles of 16 keys each ----
        v8f rs = zero8;
        const _Float16* Kc = Kb + j * 256 * 64;
#pragma unroll
        for (int t = 0; t < 16; ++t) {
            v16h b0 = load_bt_frag(Kc + (t * 16) * 64 + 0, 64);
            v16h b1 = load_bt_frag(Kc + (t * 16) * 64 + 32, 64);
            v8f c = wmma_f16(a0, b0, zero8);
            c = wmma_f16(a1, b1, c);
            const int key0 = j * 256 + t * 16 + l;
#pragma unroll
            for (int r = 0; r < 8; ++r) {
                int qn = m0 + wv * 16 + r + hf * 8;
                float s = c[r] * 0.125f;                    // 1/sqrt(64)
                s = fminf(10.f, fmaxf(-10.f, s));           // clip
                int av = adj[qn * 2048 + key0];             // adjacency mask
                s = (av == 0) ? -10.f : s;
                float p = __expf(s);                        // <= e^10: f16-safe
                rs[r] += p;
                Ps[(wv * 16 + r + hf * 8) * 256 + t * 16 + l] = (_Float16)p;
            }
        }
        // row sums: reduce over the 16 lanes of each half-group
#pragma unroll
        for (int r = 0; r < 8; ++r) {
            float v = rs[r];
            v += __shfl_xor(v, 1, 32);
            v += __shfl_xor(v, 2, 32);
            v += __shfl_xor(v, 4, 32);
            v += __shfl_xor(v, 8, 32);
            rs[r] = v;
        }

        // ---- Ptilde @ V ----
        v8f u[4];
#pragma unroll
        for (int t = 0; t < 4; ++t) u[t] = zero8;
#pragma unroll
        for (int s = 0; s < 8; ++s) {
            v16h a = load_a_frag(Ps + (wv * 16) * 256 + s * 32, 256);
#pragma unroll
            for (int t = 0; t < 4; ++t) {
                v16h bf = load_bt_frag(Vt + (t * 16) * 256 + s * 32, 256);
                u[t] = wmma_f16(a, bf, u[t]);
            }
        }
        // normalize by per-chunk softmax denominator, accumulate over chunks
#pragma unroll
        for (int r = 0; r < 8; ++r) {
            float inv = 1.0f / rs[r];
#pragma unroll
            for (int t = 0; t < 4; ++t) ctx[t][r] += u[t][r] * inv;
        }

        __syncthreads();               // all waves done reading Vt
        if (j < 7) {
            wait_asynccnt0();          // chunk j+1's V landed in Vraw
            __syncthreads();
            transpose_v();             // Vraw -> Vt
            __syncthreads();           // Vt ready; Vraw free for next issue
        }
    }

    // write context, [B,N,F] f16, col = h*64 + d
#pragma unroll
    for (int t = 0; t < 4; ++t) {
#pragma unroll
        for (int r = 0; r < 8; ++r) {
            int qn = m0 + wv * 16 + r + hf * 8;
            int rowg = bb * 2048 + qn;
            int col = h * 64 + t * 16 + l;
            CtxH[rowg * 256 + col] = (_Float16)ctx[t][r];
        }
    }
}

// ---------------------------------------------------------------------------
// Kernel 3: out = LN(ctx @ Wo^T + bo + x).  grid = 8192/16, block = 128.
// Wave w covers columns [64w, 64w+64); LN reduced across waves via LDS.
// ---------------------------------------------------------------------------
__global__ __launch_bounds__(128) void out_ln(
    const _Float16* __restrict__ CtxH, const _Float16* __restrict__ WoH,
    const float* __restrict__ bo, const float* __restrict__ x,
    const float* __restrict__ gamma, const float* __restrict__ beta,
    float* __restrict__ out) {
    __shared__ float tile[16 * 256];
    __shared__ float psum[16 * 8];
    __shared__ float psq[16 * 8];
    __shared__ float mu_s[16];
    __shared__ float rs_s[16];

    const int tid  = threadIdx.x;
    const int wv   = tid >> 5;
    const int lane = tid & 31;
    const int hf   = lane >> 4;
    const int l    = lane & 15;
    const int m0   = blockIdx.x * 16;

    v8f zero8 = {0.f, 0.f, 0.f, 0.f, 0.f, 0.f, 0.f, 0.f};
    v8f acc[4];
#pragma unroll
    for (int t = 0; t < 4; ++t) acc[t] = zero8;

#pragma unroll
    for (int s = 0; s < 8; ++s) {
        v16h a = load_a_frag(CtxH + m0 * 256 + s * 32, 256);
#pragma unroll
        for (int t = 0; t < 4; ++t) {
            int n0 = wv * 64 + t * 16;
            v16h bf = load_bt_frag(WoH + n0 * 256 + s * 32, 256);
            acc[t] = wmma_f16(a, bf, acc[t]);
        }
    }

    // bias + residual -> LDS
#pragma unroll
    for (int t = 0; t < 4; ++t) {
#pragma unroll
        for (int r = 0; r < 8; ++r) {
            int row = r + hf * 8;               // 0..15
            int col = wv * 64 + t * 16 + l;     // 0..255
            float v = acc[t][r] + bo[col] + x[(m0 + row) * 256 + col];
            tile[row * 256 + col] = v;
        }
    }
    __syncthreads();

    // LayerNorm: 8 threads per row, 32 cols each
    {
        int row = tid >> 3, seg = tid & 7;
        float s = 0.f, q = 0.f;
        for (int c = seg * 32; c < seg * 32 + 32; ++c) {
            float v = tile[row * 256 + c];
            s += v; q += v * v;
        }
        psum[row * 8 + seg] = s;
        psq[row * 8 + seg]  = q;
    }
    __syncthreads();
    if (tid < 16) {
        float s = 0.f, q = 0.f;
        for (int i = 0; i < 8; ++i) { s += psum[tid * 8 + i]; q += psq[tid * 8 + i]; }
        float mu  = s * (1.0f / 256.0f);
        float var = q * (1.0f / 256.0f) - mu * mu;
        mu_s[tid] = mu;
        rs_s[tid] = rsqrtf(var + 1e-5f);
    }
    __syncthreads();
    for (int i = tid; i < 16 * 256; i += 128) {
        int row = i >> 8, col = i & 255;
        out[(m0 + row) * 256 + col] =
            (tile[i] - mu_s[row]) * rs_s[row] * gamma[col] + beta[col];
    }
}

// ---------------------------------------------------------------------------
extern "C" void kernel_launch(void* const* d_in, const int* in_sizes, int n_in,
                              void* d_out, int out_size, void* d_ws, size_t ws_size,
                              hipStream_t stream) {
    const float* x     = (const float*)d_in[0];
    const int*   adj   = (const int*)d_in[1];
    const float* Wq    = (const float*)d_in[2];
    const float* bq    = (const float*)d_in[3];
    const float* Wk    = (const float*)d_in[4];
    const float* bk    = (const float*)d_in[5];
    const float* Wv    = (const float*)d_in[6];
    const float* bv    = (const float*)d_in[7];
    const float* Wo    = (const float*)d_in[8];
    const float* bo    = (const float*)d_in[9];
    const float* gamma = (const float*)d_in[10];
    const float* beta  = (const float*)d_in[11];

    char* ws = (char*)d_ws;
    _Float16* xh   = (_Float16*)(ws + 0);           // 8192*256
    _Float16* WqH  = (_Float16*)(ws + 4194304);     // 256*256
    _Float16* WkH  = (_Float16*)(ws + 4325376);
    _Float16* WvH  = (_Float16*)(ws + 4456448);
    _Float16* WoH  = (_Float16*)(ws + 4587520);
    _Float16* Qh   = (_Float16*)(ws + 4718592);     // 16*2048*64
    _Float16* Kh   = (_Float16*)(ws + 8912896);
    _Float16* Vh   = (_Float16*)(ws + 13107200);
    _Float16* CtxH = (_Float16*)(ws + 17301504);    // 8192*256

    cvt_f16<<<(2097152 + 255) / 256, 256, 0, stream>>>(x, xh, 2097152);
    cvt_f16<<<256, 256, 0, stream>>>(Wq, WqH, 65536);
    cvt_f16<<<256, 256, 0, stream>>>(Wk, WkH, 65536);
    cvt_f16<<<256, 256, 0, stream>>>(Wv, WvH, 65536);
    cvt_f16<<<256, 256, 0, stream>>>(Wo, WoH, 65536);

    qkv_proj<<<dim3(4, 128, 3), 128, 0, stream>>>(
        xh, WqH, WkH, WvH, bq, bk, bv, Qh, Kh, Vh);

    attn<<<dim3(32, 16), 128, 0, stream>>>(Qh, Kh, Vh, adj, CtxH);

    out_ln<<<512, 128, 0, stream>>>(CtxH, WoH, bo, x, gamma, beta, (float*)d_out);
}
